// Spatial_Weighting_78512002171304
// MI455X (gfx1250) — compile-verified, matrix operand order checked by reading, not run on hardware
//
#include <hip/hip_runtime.h>
#include <hip/hip_bf16.h>
#include <math.h>

// Problem constants (reference: B=4, T=8, H=12, N=784, D=64)
#define B_    4
#define T_    8
#define H_    12
#define N_    784
#define D_    64
#define BT_   32                     // B*T
#define NTILE 49                     // N_/16 (exact); 49 = 7*7
#define MAC   112                    // macro-tile edge = 7 tiles of 16
#define NMAC  7                      // N_/MAC (exact)
#define STEPS 48                     // 2 tensors * 12 heads * (64/32) K-steps
#define ROWPAD 36                    // staged row stride in floats (144 B, bank-friendly)
#define PITERS 80                    // power-iteration steps

typedef __attribute__((ext_vector_type(16))) __bf16 v16bf;
typedef __attribute__((ext_vector_type(8)))  float  v8f;

// s_wait_asynccnt <= n  (ASYNCcnt tracks GLOBAL_LOAD_ASYNC_TO_LDS_*)
#if defined(__has_builtin)
#if __has_builtin(__builtin_amdgcn_s_wait_asynccnt)
#define WAIT_ASYNC(n) __builtin_amdgcn_s_wait_asynccnt(n)
#else
#define WAIT_ASYNC(n) asm volatile("s_wait_asynccnt " #n ::: "memory")
#endif
#else
#define WAIT_ASYNC(n) asm volatile("s_wait_asynccnt " #n ::: "memory")
#endif

// ---------------------------------------------------------------------------
// Convert one lane's 16x32-bf16 WMMA operand share from f32 (ISA 7.12.2 layout):
// lane holds K = {koff..koff+7, koff+16..koff+23}, koff = 8*(lane>=16).
// `p` points at float [row][koff]; chunks are 16B-aligned.
// ---------------------------------------------------------------------------
__device__ __forceinline__ v16bf cvt_operand(const float* __restrict__ p) {
    float4 a = *(const float4*)(p + 0);
    float4 b = *(const float4*)(p + 4);
    float4 c = *(const float4*)(p + 16);
    float4 d = *(const float4*)(p + 20);
    v16bf r;
    r[0]  = (__bf16)a.x; r[1]  = (__bf16)a.y; r[2]  = (__bf16)a.z; r[3]  = (__bf16)a.w;
    r[4]  = (__bf16)b.x; r[5]  = (__bf16)b.y; r[6]  = (__bf16)b.z; r[7]  = (__bf16)b.w;
    r[8]  = (__bf16)c.x; r[9]  = (__bf16)c.y; r[10] = (__bf16)c.z; r[11] = (__bf16)c.w;
    r[12] = (__bf16)d.x; r[13] = (__bf16)d.y; r[14] = (__bf16)d.z; r[15] = (__bf16)d.w;
    return r;
}

// ---------------------------------------------------------------------------
// Kernel 1: W[bt] = threshold( sum_h K1_h K1_h^T + K2_h K2_h^T )
// Block = 7 waves (224 thr) computes a 112x112 macro-tile (49 WMMA tiles).
// Per K-step, 112x32 A/B f32 strips are DMA'd global->LDS with
// GLOBAL_LOAD_ASYNC_TO_LDS_B128 (double-buffered, s_wait_asynccnt-paced),
// then each wave runs 7 v_wmma_f32_16x16x32_bf16 against its A operand.
// ---------------------------------------------------------------------------
__global__ __launch_bounds__(224) void gram_threshold_kernel(
    const float* __restrict__ key1, const float* __restrict__ key2,
    float* __restrict__ W) {
    // double-buffered f32 strips: [buf][112 rows * 36 floats] = 16128 B each
    __shared__ float stageA[2][MAC * ROWPAD];
    __shared__ float stageB[2][MAC * ROWPAD];

    const int t    = threadIdx.x;        // 0..223
    const int lane = t & 31;
    const int w    = t >> 5;             // wave 0..6 -> output row-tile in macro
    const int bt   = blockIdx.y;
    const int bi   = blockIdx.x / NMAC;  // macro row  0..6
    const int bj   = blockIdx.x % NMAC;  // macro col  0..6

    const size_t headStride = (size_t)N_ * D_;
    const float* k1base = key1 + (size_t)bt * H_ * headStride;
    const float* k2base = key2 + (size_t)bt * H_ * headStride;

    // staging role: thread t stages one 32-float row chunk (8 x b128)
    const bool isB  = (t >= MAC);
    const int  srow = isB ? (t - MAC) : t;                  // row within strip
    const int  rowG = (isB ? bj : bi) * MAC + srow;         // global row index
    float* const ldsRow0 = (isB ? &stageB[0][0] : &stageA[0][0]) + srow * ROWPAD;
    float* const ldsRow1 = (isB ? &stageB[1][0] : &stageA[1][0]) + srow * ROWPAD;

    auto issue_stage = [&](int k) {
        const int s  = k / 24;           // tensor select
        const int h  = (k % 24) >> 1;    // head
        const int ks = k & 1;            // K half of D=64
        const float* gsrc = (s ? k2base : k1base)
                          + (size_t)h * headStride + (size_t)rowG * D_ + ks * 32;
        const uint64_t g  = (uint64_t)(uintptr_t)gsrc;
        const unsigned l  = (unsigned)(uintptr_t)((k & 1) ? ldsRow1 : ldsRow0);
        // NOTE: buffer parity == k&1 (ks alternates each step, so parity == step parity)
        #pragma unroll
        for (int c = 0; c < 8; ++c) {    // 8 x 16B = 32 floats
            asm volatile("global_load_async_to_lds_b128 %0, %1, off"
                         :: "v"(l + (unsigned)(c * 16)), "v"(g + (uint64_t)(c * 16))
                         : "memory");
        }
    };

    const int lrow = lane & 15;          // row within 16-tile (A) / col (B)
    const int koff = (lane >> 4) * 8;    // per-lane K offset (floats)

    v8f acc[NMAC];
    #pragma unroll
    for (int j = 0; j < NMAC; ++j) acc[j] = (v8f){};

    issue_stage(0);
    for (int k = 0; k < STEPS; ++k) {
        const int buf = k & 1;
        if (k + 1 < STEPS) {
            issue_stage(k + 1);
            WAIT_ASYNC(8);               // batch k done; batch k+1 in flight
        } else {
            WAIT_ASYNC(0);
        }
        __syncthreads();                 // all waves' staging of step k visible

        // A operand for this wave's row-tile
        v16bf a = cvt_operand(&stageA[buf][(w * 16 + lrow) * ROWPAD + koff]);
        #pragma unroll
        for (int j = 0; j < NMAC; ++j) {
            v16bf b = cvt_operand(&stageB[buf][(j * 16 + lrow) * ROWPAD + koff]);
            acc[j] = __builtin_amdgcn_wmma_f32_16x16x32_bf16(
                         false, a, false, b, (short)0, acc[j], false, false);
        }
        __syncthreads();                 // close readers before buffer reuse
    }

    // C/D layout: VGPR r -> M = r + 8*(lane>=16), N = lane&15
    float* Wt = W + (size_t)bt * N_ * N_;
    const int m0 = bi * MAC + w * 16 + (lane >> 4) * 8;
    #pragma unroll
    for (int j = 0; j < NMAC; ++j) {
        const int n = bj * MAC + j * 16 + lrow;
        #pragma unroll
        for (int r = 0; r < 8; ++r) {
            float v = (acc[j][r] > 0.2f) ? 1.0f : 1e-5f;
            Wt[(size_t)(m0 + r) * N_ + n] = v;
        }
    }
}

// ---------------------------------------------------------------------------
// Block reductions (wave32, blockDim == 1024 -> 32 waves)
// ---------------------------------------------------------------------------
__device__ __forceinline__ float wave_sum(float v) {
    #pragma unroll
    for (int o = 16; o; o >>= 1) v += __shfl_xor(v, o, 32);
    return v;
}
__device__ __forceinline__ float wave_max(float v) {
    #pragma unroll
    for (int o = 16; o; o >>= 1) v = fmaxf(v, __shfl_xor(v, o, 32));
    return v;
}
__device__ __forceinline__ float block_sum(float v, float* red, int tid) {
    const int lane = tid & 31, wid = tid >> 5;
    v = wave_sum(v);
    if (lane == 0) red[wid] = v;
    __syncthreads();
    if (wid == 0) {
        float t = wave_sum(red[lane]);
        if (lane == 0) red[0] = t;
    }
    __syncthreads();
    float r = red[0];
    __syncthreads();
    return r;
}
__device__ __forceinline__ float block_max(float v, float* red, int tid) {
    const int lane = tid & 31, wid = tid >> 5;
    v = wave_max(v);
    if (lane == 0) red[wid] = v;
    __syncthreads();
    if (wid == 0) {
        float t = wave_max(red[lane]);
        if (lane == 0) red[0] = t;
    }
    __syncthreads();
    float r = red[0];
    __syncthreads();
    return r;
}

// ---------------------------------------------------------------------------
// Kernel 2: per (b,t) — degrees, deflated power iteration on M = I + D^-1/2 W D^-1/2
// (top eigvec u0 = normalized sqrt(d) deflated -> Fiedler vector), then
// bipartition + masked softmax. W (78.7 MB) is L2-resident on MI455X (192 MB L2).
// ---------------------------------------------------------------------------
__global__ __launch_bounds__(1024) void fiedler_softmax_kernel(
    const float* __restrict__ W, float* __restrict__ out) {
    const int bt   = blockIdx.x;
    const float* Wm = W + (size_t)bt * N_ * N_;
    const int tid  = threadIdx.x;
    const int lane = tid & 31;
    const int wid  = tid >> 5;          // 0..31

    __shared__ float isd[N_];           // 1/sqrt(d)
    __shared__ float u0 [N_];           // normalized sqrt(d)
    __shared__ float xv [N_];           // iterate
    __shared__ float xs [N_];           // scaled iterate / scratch
    __shared__ float red[32];

    // --- degrees (wave per row, coalesced) ---
    for (int n = wid; n < N_; n += 32) {
        const float* row = Wm + (size_t)n * N_;
        float s = 0.f;
        for (int m = lane; m < N_; m += 32) s += row[m];
        s = wave_sum(s);
        if (lane == 0) isd[n] = rsqrtf(s);
    }
    __syncthreads();

    // --- u0 = sqrt(d) / ||sqrt(d)|| ---
    float pd = 0.f;
    for (int n = tid; n < N_; n += 1024) { float r = isd[n]; pd += 1.f / (r * r); }
    float totd = block_sum(pd, red, tid);
    float invn = rsqrtf(totd);
    for (int n = tid; n < N_; n += 1024) u0[n] = invn / isd[n];

    // --- init x: linear ramp (generic, non-degenerate) ---
    for (int n = tid; n < N_; n += 1024) xv[n] = (float)n - 0.5f * (float)(N_ - 1);
    __syncthreads();

    // --- deflated power iteration ---
    for (int it = 0; it < PITERS; ++it) {
        for (int n = tid; n < N_; n += 1024) xs[n] = isd[n] * xv[n];
        __syncthreads();
        // y = isd .* (W @ xs) + x     (M = I + What)
        for (int n = wid; n < N_; n += 32) {
            const float* row = Wm + (size_t)n * N_;
            if (n + 32 < N_) __builtin_prefetch(row + (size_t)32 * N_, 0, 1);
            float s = 0.f;
            for (int m = lane; m < N_; m += 32) s += row[m] * xs[m];
            s = wave_sum(s);
            if (lane == 0) xv[n] = isd[n] * s + xv[n];
        }
        __syncthreads();
        // deflate u0, renormalize
        float p1 = 0.f;
        for (int n = tid; n < N_; n += 1024) p1 += xv[n] * u0[n];
        float s1 = block_sum(p1, red, tid);
        float p2 = 0.f;
        for (int n = tid; n < N_; n += 1024) {
            float y = xv[n] - s1 * u0[n];
            xv[n] = y;
            p2 += y * y;
        }
        float nrm = block_sum(p2, red, tid);
        float sc = rsqrtf(nrm);
        for (int n = tid; n < N_; n += 1024) xv[n] *= sc;
        __syncthreads();
    }

    // --- v = u1 .* isd ; mean ---
    float pm = 0.f;
    for (int n = tid; n < N_; n += 1024) {
        float v = xv[n] * isd[n];
        xs[n] = v;
        pm += v;
    }
    float avg = block_sum(pm, red, tid) * (1.0f / (float)N_);
    const bool flip = !(avg > 0.f);

    // --- bipartition mask + masked |v| (0 -> -inf) ---
    float pmax = -INFINITY;
    for (int n = tid; n < N_; n += 1024) {
        float v = xs[n];
        bool bip = flip ? !(v > 0.f) : (v > 0.f);
        float e = (bip && v != 0.f) ? fabsf(v) : -INFINITY;
        xv[n] = e;
        pmax = fmaxf(pmax, e);
    }
    float emax = block_max(pmax, red, tid);

    // --- softmax over n ---
    float ps = 0.f;
    for (int n = tid; n < N_; n += 1024) {
        float e = xv[n];
        float w = (e == -INFINITY) ? 0.f : expf(e - emax);
        xs[n] = w;
        ps += w;
    }
    float tot = block_sum(ps, red, tid);
    float inv = 1.0f / tot;
    float* o = out + (size_t)bt * N_;
    for (int n = tid; n < N_; n += 1024) o[n] = xs[n] * inv;
}

// ---------------------------------------------------------------------------
extern "C" void kernel_launch(void* const* d_in, const int* in_sizes, int n_in,
                              void* d_out, int out_size, void* d_ws, size_t ws_size,
                              hipStream_t stream) {
    const float* key1 = (const float*)d_in[0];   // (32, 12, 784, 64) f32
    const float* key2 = (const float*)d_in[1];   // (32, 12, 784, 64) f32
    float* W   = (float*)d_ws;                   // 32*784*784 f32 = 78.7 MB scratch
    float* out = (float*)d_out;                  // (4, 8, 784) f32

    gram_threshold_kernel<<<dim3(NMAC * NMAC, BT_), 224, 0, stream>>>(key1, key2, W);
    fiedler_softmax_kernel<<<BT_, 1024, 0, stream>>>(W, out);
}
